// KMeansTokenizer_91061896610269
// MI455X (gfx1250) — matrix operand order repaired and checked by reading, not compiled, not run on hardware
//
#include <hip/hip_runtime.h>

// ---------------------------------------------------------------------------
// KMeans tokenizer for MI455X (gfx1250, wave32, WMMA).
//   tokens[n] = argmin_k ||patch_n - vocab_k||^2
//             = argmax_k ( patch_n . vocab_k  -  0.5*||vocab_k||^2 )
// GEMM (32768 x 8192 x 64) on v_wmma_f32_16x16x32_bf16 with bf16x3 hi/lo
// split (hi*hi + hi*lo + lo*hi, error ~2^-16 => fp32-faithful argmin).
// -0.5*||c||^2 is folded into the WMMA accumulator initialization.
//
// Register blocking: each wave owns TWO 16-row A tiles (32 patch rows), so
// every 4KB B-tile load feeds 12 WMMAs (halves L2 traffic vs 1 tile/wave).
// ---------------------------------------------------------------------------

typedef __attribute__((ext_vector_type(16))) __bf16 v16bf;
typedef __attribute__((ext_vector_type(8)))  float  v8f;

#define VQ_VOCAB 8192
#define VQ_DIM   64
#define VQ_ROWS  32768          // 16 * 2048 patches
#define VQ_TILES (VQ_VOCAB / 16)
#define VQ_MTILES 2             // 16-row A tiles per wave

static __device__ __forceinline__ __bf16 f32_to_bf16_rne(float x) {
  unsigned u = __builtin_bit_cast(unsigned, x);
  u += 0x7FFFu + ((u >> 16) & 1u);            // round to nearest even
  unsigned short h = (unsigned short)(u >> 16);
  return __builtin_bit_cast(__bf16, h);
}
static __device__ __forceinline__ float bf16_to_f32(__bf16 h) {
  unsigned u = ((unsigned)__builtin_bit_cast(unsigned short, h)) << 16;
  return __builtin_bit_cast(float, u);
}

// ---------------------------------------------------------------------------
// Prep kernel: one thread per vocab row.
//   negHalfC2[k] = -0.5 * ||vocab_k||^2                  (f32, exact)
//   hiB/loB[k][i] = packed pair of bf16 (dims 2i, 2i+1)  (hi / residual-lo)
// Row-major packing matches the B-fragment element order (K ascending).
// ---------------------------------------------------------------------------
__global__ void __launch_bounds__(256)
vq_prep_kernel(const float* __restrict__ vocab,
               float* __restrict__ negHalfC2,
               unsigned* __restrict__ hiB,
               unsigned* __restrict__ loB) {
  const int r = blockIdx.x * blockDim.x + threadIdx.x;
  if (r >= VQ_VOCAB) return;
  const float* row = vocab + (size_t)r * VQ_DIM;
  float c2 = 0.0f;
#pragma unroll
  for (int i = 0; i < VQ_DIM / 2; ++i) {
    const float x0 = row[2 * i];
    const float x1 = row[2 * i + 1];
    c2 = fmaf(x0, x0, c2);
    c2 = fmaf(x1, x1, c2);
    const __bf16 h0 = f32_to_bf16_rne(x0);
    const __bf16 h1 = f32_to_bf16_rne(x1);
    const __bf16 l0 = f32_to_bf16_rne(x0 - bf16_to_f32(h0));
    const __bf16 l1 = f32_to_bf16_rne(x1 - bf16_to_f32(h1));
    const unsigned hp = ((unsigned)__builtin_bit_cast(unsigned short, h1) << 16) |
                        (unsigned)__builtin_bit_cast(unsigned short, h0);
    const unsigned lp = ((unsigned)__builtin_bit_cast(unsigned short, l1) << 16) |
                        (unsigned)__builtin_bit_cast(unsigned short, l0);
    hiB[(size_t)r * (VQ_DIM / 2) + i] = hp;
    loB[(size_t)r * (VQ_DIM / 2) + i] = lp;
  }
  negHalfC2[r] = -0.5f * c2;
}

// ---------------------------------------------------------------------------
// Main kernel: one wave per 32 patch rows (two 16x64 A tiles), 1024 waves.
// Per 16-code tile: load B once; 12x v_wmma_f32_16x16x32_bf16 (6 per A tile,
// interleaved across the two accumulators); running per-lane argmax; final
// shfl_xor reduction across the 16 columns.
//
// Fragment layouts used here (16-bit, wave32, contiguous K = 16h + s per chunk):
//   A: lane L -> row L%16 ; elems 0..7  = K (2*lo8..2*lo8+7),
//                          elems 8..15 = K (2*lo8+8 .. 2*lo8+15), lo8 = (L>=16)*8
//   B: lane L -> col L%16 ; elems j = K (h*16 + j), h = L>>4 (per 32-K chunk)
//   C/D: VGPR r -> (row r, col lane) lanes 0-15 ; (row r+8, col lane-16) else
// ---------------------------------------------------------------------------
__global__ void __launch_bounds__(256)
vq_argmin_kernel(const float* __restrict__ patches,
                 const float* __restrict__ negHalfC2,
                 const unsigned* __restrict__ hiB,
                 const unsigned* __restrict__ loB,
                 int* __restrict__ tokens) {
  const int lane    = threadIdx.x & 31;
  const int wave    = threadIdx.x >> 5;
  const int gwave   = blockIdx.x * (blockDim.x >> 5) + wave;
  const int rowBase = gwave * (16 * VQ_MTILES);
  const int m       = lane & 15;   // row (A) / col (B,C,D) within tile
  const int h       = lane >> 4;   // lane-half selector
  const int lo8     = h * 8;

  // ---- Build A fragments once: bf16 hi/lo split of 2x(16x64) patch tiles.
  v16bf aHi[VQ_MTILES][2], aLo[VQ_MTILES][2];
#pragma unroll
  for (int u = 0; u < VQ_MTILES; ++u) {
    const float* arow = patches + (size_t)(rowBase + u * 16 + m) * VQ_DIM;
#pragma unroll
    for (int c = 0; c < 2; ++c) {
#pragma unroll
      for (int j = 0; j < 8; ++j) {
        const float x = arow[c * 32 + 2 * lo8 + j];
        const __bf16 hx = f32_to_bf16_rne(x);
        aHi[u][c][j] = hx;
        aLo[u][c][j] = f32_to_bf16_rne(x - bf16_to_f32(hx));
        const float y = arow[c * 32 + 2 * lo8 + 8 + j];
        const __bf16 hy = f32_to_bf16_rne(y);
        aHi[u][c][8 + j] = hy;
        aLo[u][c][8 + j] = f32_to_bf16_rne(y - bf16_to_f32(hy));
      }
    }
  }

  float best[VQ_MTILES][8];
  int   bidx[VQ_MTILES][8];
#pragma unroll
  for (int u = 0; u < VQ_MTILES; ++u)
#pragma unroll
    for (int r = 0; r < 8; ++r) { best[u][r] = -3.4e38f; bidx[u][r] = 0; }

  const int ku = h * 8;  // uint offset of this lane-half's K slice in a chunk

  for (int t = 0; t < VQ_TILES; ++t) {
    const int code = t * 16 + m;
    const unsigned* hp = hiB + (size_t)code * (VQ_DIM / 2) + ku;
    const unsigned* lp = loB + (size_t)code * (VQ_DIM / 2) + ku;

    // Each v16bf = 32B, 32B-aligned (row stride 128B, ku*4 in {0,32}).
    const v16bf bh0 = *(const v16bf*)(hp);        // K chunk 0 (dims  0..31)
    const v16bf bh1 = *(const v16bf*)(hp + 16);   // K chunk 1 (dims 32..63)
    const v16bf bl0 = *(const v16bf*)(lp);
    const v16bf bl1 = *(const v16bf*)(lp + 16);

    if (t + 1 < VQ_TILES) {                       // next tile: +16 rows = +512 uints
      __builtin_prefetch((const void*)(hp + 512), 0, 1);
      __builtin_prefetch((const void*)(lp + 512), 0, 1);
    }

    const float nc2 = negHalfC2[code];            // fold -0.5||c||^2 into C
    v8f acc[VQ_MTILES];
#pragma unroll
    for (int u = 0; u < VQ_MTILES; ++u)
      acc[u] = (v8f){ nc2, nc2, nc2, nc2, nc2, nc2, nc2, nc2 };

    // bf16x3: hi*hi + hi*lo + lo*hi. Interleave the two accumulator chains
    // so consecutive WMMAs are independent (better XDL pipelining).
#pragma unroll
    for (int u = 0; u < VQ_MTILES; ++u)
      acc[u] = __builtin_amdgcn_wmma_f32_16x16x32_bf16(false, aHi[u][0], false, bh0, (short)0, acc[u], false, false);
#pragma unroll
    for (int u = 0; u < VQ_MTILES; ++u)
      acc[u] = __builtin_amdgcn_wmma_f32_16x16x32_bf16(false, aHi[u][0], false, bl0, (short)0, acc[u], false, false);
#pragma unroll
    for (int u = 0; u < VQ_MTILES; ++u)
      acc[u] = __builtin_amdgcn_wmma_f32_16x16x32_bf16(false, aLo[u][0], false, bh0, (short)0, acc[u], false, false);
#pragma unroll
    for (int u = 0; u < VQ_MTILES; ++u)
      acc[u] = __builtin_amdgcn_wmma_f32_16x16x32_bf16(false, aHi[u][1], false, bh1, (short)0, acc[u], false, false);
#pragma unroll
    for (int u = 0; u < VQ_MTILES; ++u)
      acc[u] = __builtin_amdgcn_wmma_f32_16x16x32_bf16(false, aHi[u][1], false, bl1, (short)0, acc[u], false, false);
#pragma unroll
    for (int u = 0; u < VQ_MTILES; ++u)
      acc[u] = __builtin_amdgcn_wmma_f32_16x16x32_bf16(false, aLo[u][1], false, bh1, (short)0, acc[u], false, false);

#pragma unroll
    for (int u = 0; u < VQ_MTILES; ++u)
#pragma unroll
      for (int r = 0; r < 8; ++r) {
        if (acc[u][r] > best[u][r]) { best[u][r] = acc[u][r]; bidx[u][r] = code; }
      }
  }

  // ---- Argmax across the 16 columns held by each 16-lane group.
  // Ties -> smaller index (matches argmin first-occurrence).
#pragma unroll
  for (int u = 0; u < VQ_MTILES; ++u) {
#pragma unroll
    for (int r = 0; r < 8; ++r) {
      float s = best[u][r];
      int   i = bidx[u][r];
#pragma unroll
      for (int mask = 8; mask >= 1; mask >>= 1) {
        const float os = __shfl_xor(s, mask, 16);
        const int   oi = __shfl_xor(i, mask, 16);
        if (os > s || (os == s && oi < i)) { s = os; i = oi; }
      }
      best[u][r] = s;
      bidx[u][r] = i;
    }
    // Lane 0 holds rows +0..7 ; lane 16 holds rows +8..15 of this tile.
    if (m == 0) {
      const int outRow = rowBase + u * 16 + h * 8;
#pragma unroll
      for (int r = 0; r < 8; ++r) tokens[outRow + r] = bidx[u][r];
    }
  }
}

// ---------------------------------------------------------------------------
// Workspace layout (needs 32KB + 1MB + 1MB = 2,129,920 bytes of d_ws):
//   [0      , 32KB)          negHalfC2 : f32[8192]
//   [32KB   , 32KB+1MB)      hiB       : u32[8192*32]  (packed bf16 hi pairs)
//   [32KB+1M, 32KB+2MB)      loB       : u32[8192*32]  (packed bf16 lo pairs)
// ---------------------------------------------------------------------------
extern "C" void kernel_launch(void* const* d_in, const int* in_sizes, int n_in,
                              void* d_out, int out_size, void* d_ws, size_t ws_size,
                              hipStream_t stream) {
  (void)in_sizes; (void)n_in; (void)out_size; (void)ws_size;
  const float* patches = (const float*)d_in[0];   // [16*2048, 64] f32
  const float* vocab   = (const float*)d_in[1];   // [8192, 64]    f32
  int* tokens = (int*)d_out;                      // [16*2048]     int32 (reference dtype)

  char* ws = (char*)d_ws;
  float*    negHalfC2 = (float*)ws;
  unsigned* hiB       = (unsigned*)(ws + 32768);
  unsigned* loB       = (unsigned*)(ws + 32768 + (size_t)VQ_VOCAB * VQ_DIM * 2);

  vq_prep_kernel<<<VQ_VOCAB / 256, 256, 0, stream>>>(vocab, negHalfC2, hiB, loB);

  // 1024 waves (32 rows each) = 128 blocks x 256 threads (8 waves).
  vq_argmin_kernel<<<(VQ_ROWS / (16 * VQ_MTILES)) / 8, 256, 0, stream>>>(
      patches, negHalfC2, hiB, loB, tokens);
}
